// MoeLayer_19000935318064
// MI455X (gfx1250) — matrix operand order repaired and checked by reading, not compiled, not run on hardware
//
#include <hip/hip_runtime.h>
#include <cmath>

// ---------------- problem constants (from reference) ----------------
constexpr int kS   = 4096;   // B*L tokens
constexpr int kD   = 1024;   // model dim
constexpr int kH   = 4096;   // hidden dim
constexpr int kE   = 8;      // experts
constexpr int kCAP = 1280;   // capacity per expert

// ---------------- WMMA types ----------------
typedef __bf16 v16bf __attribute__((ext_vector_type(16)));
typedef __bf16 v8bf  __attribute__((ext_vector_type(8)));
typedef __bf16 v4bf  __attribute__((ext_vector_type(4)));
typedef float  v8f   __attribute__((ext_vector_type(8)));

static __device__ __forceinline__ v8f zero8() {
  v8f z = {0.f, 0.f, 0.f, 0.f, 0.f, 0.f, 0.f, 0.f};
  return z;
}

static __device__ __forceinline__ v16bf cat8(v8bf a, v8bf b) {
  return __builtin_shufflevector(a, b, 0, 1, 2, 3, 4, 5, 6, 7,
                                 8, 9, 10, 11, 12, 13, 14, 15);
}

static __device__ __forceinline__ float gelu_tanh(float x) {
  // jax.nn.gelu (approximate=True)
  float x3 = x * x * x;
  return 0.5f * x * (1.0f + tanhf(0.7978845608028654f * (x + 0.044715f * x3)));
}

// CDNA5 async memory->LDS copy (16B per lane), tracked by ASYNCcnt.
static __device__ __forceinline__ void async_copy_b128(uint32_t lds_off,
                                                       const void* gptr) {
  asm volatile("global_load_async_to_lds_b128 %0, %1, off"
               :: "v"(lds_off), "v"((unsigned long long)(uintptr_t)gptr)
               : "memory");
}
static __device__ __forceinline__ void wait_async0() {
  asm volatile("s_wait_asynccnt 0" ::: "memory");
}
static __device__ __forceinline__ uint32_t lds_addr(const void* p) {
  // generic LDS pointer low 32 bits == wave-relative LDS byte offset
  return (uint32_t)(uintptr_t)p;
}

// =====================================================================
// 0) Weight pre-pass: W[E][K][N] fp32 -> transposed bf16 hi/lo planes
//    T[E][N][K].  32x32 LDS tile transpose; both directions coalesced.
// =====================================================================
__global__ __launch_bounds__(256) void split_transpose_kernel(
    const float* __restrict__ W, __bf16* __restrict__ Thi,
    __bf16* __restrict__ Tlo, int K, int N) {
  __shared__ float tile[32][33];
  const float* We = W + (size_t)blockIdx.z * K * N;
  __bf16* Thie = Thi + (size_t)blockIdx.z * K * N;
  __bf16* Tloe = Tlo + (size_t)blockIdx.z * K * N;
  int k0 = blockIdx.x * 32, n0 = blockIdx.y * 32;
  int tid = threadIdx.x;
#pragma unroll
  for (int p = 0; p < 4; ++p) {
    int idx = tid + p * 256;
    int r = idx >> 5, c = idx & 31;
    tile[r][c] = We[(size_t)(k0 + r) * N + n0 + c];
  }
  __syncthreads();
#pragma unroll
  for (int p = 0; p < 4; ++p) {
    int idx = tid + p * 256;
    int r = idx >> 5, c = idx & 31;
    float x = tile[c][r];
    __bf16 hh = (__bf16)x;
    Thie[(size_t)(n0 + r) * K + k0 + c] = hh;
    Tloe[(size_t)(n0 + r) * K + k0 + c] = (__bf16)(x - (float)hh);
  }
}

// =====================================================================
// 1) Gate kernel: logits = tokens @ gw^T, softmax, top1/top2.
// =====================================================================
__global__ __launch_bounds__(256) void gate_kernel(
    const float* __restrict__ tokens, const float* __restrict__ gw,
    float* __restrict__ gates_all, int* __restrict__ top1, int* __restrict__ top2,
    float* __restrict__ g1v, float* __restrict__ g2v) {
  int wave = threadIdx.x >> 5;
  int lane = threadIdx.x & 31;
  int s = blockIdx.x * 8 + wave;
  const float* t = tokens + (size_t)s * kD;

  float acc[kE];
#pragma unroll
  for (int e = 0; e < kE; ++e) acc[e] = 0.f;
  for (int k = lane; k < kD; k += 32) {
    float tv = t[k];
#pragma unroll
    for (int e = 0; e < kE; ++e) acc[e] += tv * gw[e * kD + k];
  }
#pragma unroll
  for (int off = 16; off > 0; off >>= 1) {
#pragma unroll
    for (int e = 0; e < kE; ++e) acc[e] += __shfl_xor(acc[e], off, 32);
  }

  if (lane == 0) {
    float mx = acc[0];
#pragma unroll
    for (int e = 1; e < kE; ++e) mx = fmaxf(mx, acc[e]);
    float g[kE];
    float sum = 0.f;
#pragma unroll
    for (int e = 0; e < kE; ++e) { g[e] = expf(acc[e] - mx); sum += g[e]; }
    float inv = 1.0f / sum;
#pragma unroll
    for (int e = 0; e < kE; ++e) { g[e] *= inv; gates_all[s * kE + e] = g[e]; }

    int i1 = 0; float b1 = g[0];
#pragma unroll
    for (int e = 1; e < kE; ++e) if (g[e] > b1) { b1 = g[e]; i1 = e; }
    int i2 = 0; float b2 = -1.f;
#pragma unroll
    for (int e = 0; e < kE; ++e) if (e != i1 && g[e] > b2) { b2 = g[e]; i2 = e; }
    top1[s] = i1; top2[s] = i2; g1v[s] = b1; g2v[s] = b2;
  }
}

// =====================================================================
// 2) Scan kernel: deterministic routing (see round-1 notes).
// =====================================================================
__global__ __launch_bounds__(1024) void scan_kernel(
    const int* __restrict__ top1, const int* __restrict__ top2,
    const float* __restrict__ g1v, const float* __restrict__ g2v,
    const float* __restrict__ gates_all,
    int* __restrict__ pos1, int* __restrict__ pos2,
    float* __restrict__ wgt1, float* __restrict__ wgt2,
    int* __restrict__ slot_token, float* __restrict__ out_laux) {
  __shared__ unsigned long long sc0[1024];
  __shared__ unsigned long long sc1[1024];
  __shared__ float red[1024];
  int tid = threadIdx.x;

  for (int i = tid; i < kE * kCAP; i += 1024) slot_token[i] = -1;

  int s0 = tid * 4;
  int e1[4], e2[4];
#pragma unroll
  for (int j = 0; j < 4; ++j) { e1[j] = top1[s0 + j]; e2[j] = top2[s0 + j]; }

  unsigned long long c0 = 0ull, c1 = 0ull;
#pragma unroll
  for (int j = 0; j < 4; ++j) {
    if (e1[j] < 4) c0 += 1ull << (16 * e1[j]); else c1 += 1ull << (16 * (e1[j] - 4));
  }
  sc0[tid] = c0; sc1[tid] = c1;
  __syncthreads();
  for (int off = 1; off < 1024; off <<= 1) {
    unsigned long long a0 = 0ull, a1 = 0ull;
    if (tid >= off) { a0 = sc0[tid - off]; a1 = sc1[tid - off]; }
    __syncthreads();
    sc0[tid] += a0; sc1[tid] += a1;
    __syncthreads();
  }
  unsigned long long ex0 = sc0[tid] - c0, ex1 = sc1[tid] - c1;
  unsigned long long t0 = sc0[1023], t1 = sc1[1023];
  int run[kE], tot1[kE];
#pragma unroll
  for (int e = 0; e < 4; ++e) {
    run[e]      = (int)((ex0 >> (16 * e)) & 0xFFFFull);
    run[e + 4]  = (int)((ex1 >> (16 * e)) & 0xFFFFull);
    tot1[e]     = (int)((t0 >> (16 * e)) & 0xFFFFull);
    tot1[e + 4] = (int)((t1 >> (16 * e)) & 0xFFFFull);
  }
  bool keep1[4];
#pragma unroll
  for (int j = 0; j < 4; ++j) {
    int e = e1[j];
    int p = run[e]++;
    keep1[j] = (p < kCAP);
    pos1[s0 + j] = keep1[j] ? p : -1;
    if (keep1[j]) slot_token[e * kCAP + p] = s0 + j;
  }
  __syncthreads();

  c0 = 0ull; c1 = 0ull;
#pragma unroll
  for (int j = 0; j < 4; ++j) {
    if (e2[j] < 4) c0 += 1ull << (16 * e2[j]); else c1 += 1ull << (16 * (e2[j] - 4));
  }
  sc0[tid] = c0; sc1[tid] = c1;
  __syncthreads();
  for (int off = 1; off < 1024; off <<= 1) {
    unsigned long long a0 = 0ull, a1 = 0ull;
    if (tid >= off) { a0 = sc0[tid - off]; a1 = sc1[tid - off]; }
    __syncthreads();
    sc0[tid] += a0; sc1[tid] += a1;
    __syncthreads();
  }
  ex0 = sc0[tid] - c0; ex1 = sc1[tid] - c1;
  int run2[kE];
#pragma unroll
  for (int e = 0; e < 4; ++e) {
    run2[e]     = (int)((ex0 >> (16 * e)) & 0xFFFFull);
    run2[e + 4] = (int)((ex1 >> (16 * e)) & 0xFFFFull);
  }
  bool keep2[4];
#pragma unroll
  for (int j = 0; j < 4; ++j) {
    int e = e2[j];
    int p = tot1[e] + run2[e]++;
    keep2[j] = (p < kCAP);
    pos2[s0 + j] = keep2[j] ? p : -1;
    if (keep2[j]) slot_token[e * kCAP + p] = s0 + j;
  }

#pragma unroll
  for (int j = 0; j < 4; ++j) {
    float g1 = keep1[j] ? g1v[s0 + j] : 0.f;
    float g2 = keep2[j] ? g2v[s0 + j] : 0.f;
    float den = fmaxf(g1 + g2, 1.1920929e-07f);
    wgt1[s0 + j] = g1 / den;
    wgt2[s0 + j] = g2 / den;
  }

  float m[kE];
#pragma unroll
  for (int e = 0; e < kE; ++e) m[e] = 0.f;
#pragma unroll
  for (int j = 0; j < 4; ++j)
#pragma unroll
    for (int e = 0; e < kE; ++e) m[e] += gates_all[(s0 + j) * kE + e];

  float laux_acc = 0.f;
  for (int e = 0; e < kE; ++e) {
    __syncthreads();
    red[tid] = m[e];
    __syncthreads();
    for (int off = 512; off > 0; off >>= 1) {
      if (tid < off) red[tid] += red[tid + off];
      __syncthreads();
    }
    if (tid == 0) laux_acc += red[0] * (float)tot1[e];
  }
  if (tid == 0)
    out_laux[0] = (float)kE * laux_acc / ((float)kS * (float)kS);
}

// =====================================================================
// 3) Gather (dispatch): write bf16 hi/lo planes of dispatched tokens.
// =====================================================================
__global__ __launch_bounds__(256) void gather_kernel(
    const float* __restrict__ tokens, const int* __restrict__ slot_token,
    __bf16* __restrict__ dhi, __bf16* __restrict__ dlo) {
  int slot = blockIdx.x;
  int s = slot_token[slot];
  float4 v = {0.f, 0.f, 0.f, 0.f};
  if (s >= 0) v = ((const float4*)tokens)[(size_t)s * (kD / 4) + threadIdx.x];
  float xs[4] = {v.x, v.y, v.z, v.w};
  v4bf h, l;
#pragma unroll
  for (int c = 0; c < 4; ++c) {
    __bf16 hh = (__bf16)xs[c];
    h[c] = hh;
    l[c] = (__bf16)(xs[c] - (float)hh);
  }
  ((v4bf*)dhi)[(size_t)slot * (kD / 4) + threadIdx.x] = h;
  ((v4bf*)dlo)[(size_t)slot * (kD / 4) + threadIdx.x] = l;
}

// =====================================================================
// 4) Batched-expert GEMM, all-bf16 operands (pre-split hi/lo planes),
//    bf16x3 WMMA; zero conversion VALU in the inner loop.
//    A[M][K] hi/lo row-major; Bt[N][K] hi/lo row-major (weights^T).
//    Block tile 128x128, 8 waves (4x2), 32x64 per wave (2x4 WMMA tiles).
//    Global->LDS via GLOBAL_LOAD_ASYNC_TO_LDS_B128, double-buffered.
// =====================================================================
constexpr int LDT = 40;  // padded row stride (bf16) for all 128x32 tiles

template <bool DO_GELU, bool SPLIT_OUT>
__global__ __launch_bounds__(256) void gemm_bf16x3_kernel(
    const __bf16* __restrict__ Ahi, const __bf16* __restrict__ Alo,
    const __bf16* __restrict__ Bthi, const __bf16* __restrict__ Btlo,
    const float* __restrict__ bias, float* __restrict__ Cf,
    __bf16* __restrict__ Chi, __bf16* __restrict__ Clo,
    int M, int N, int K) {
  __shared__ __bf16 sAh[2][128 * LDT];
  __shared__ __bf16 sAl[2][128 * LDT];
  __shared__ __bf16 sBh[2][128 * LDT];
  __shared__ __bf16 sBl[2][128 * LDT];

  const __bf16* Ahe = Ahi + (size_t)blockIdx.z * M * K;
  const __bf16* Ale = Alo + (size_t)blockIdx.z * M * K;
  const __bf16* Bhe = Bthi + (size_t)blockIdx.z * N * K;
  const __bf16* Ble = Btlo + (size_t)blockIdx.z * N * K;
  const float* be = bias + (size_t)blockIdx.z * N;

  int m0 = blockIdx.x * 128;
  int n0 = blockIdx.y * 128;
  int tid = threadIdx.x;

  int lane = tid & 31;
  int wid  = tid >> 5;
  int wm = (wid & 3) * 32;    // wave row offset in block tile
  int wn = (wid >> 2) * 64;   // wave col offset in block tile
  int l16 = lane & 15;
  int lhi = lane >> 4;        // 0 or 1

  // stage K-chunk k0 into buffer b: 4 tiles x 8KB, 8 async b128 per thread
  auto stage = [&](int k0, int b) {
#pragma unroll
    for (int p = 0; p < 2; ++p) {
      int idx = tid + p * 256;
      int r = idx >> 2, sg = (idx & 3) * 8;  // 8 bf16 = 16B segment
      size_t ga = (size_t)(m0 + r) * K + k0 + sg;
      size_t gb = (size_t)(n0 + r) * K + k0 + sg;
      uint32_t lo = (uint32_t)(r * LDT + sg) * 2u;
      async_copy_b128(lds_addr(&sAh[b][0]) + lo, Ahe + ga);
      async_copy_b128(lds_addr(&sAl[b][0]) + lo, Ale + ga);
      async_copy_b128(lds_addr(&sBh[b][0]) + lo, Bhe + gb);
      async_copy_b128(lds_addr(&sBl[b][0]) + lo, Ble + gb);
    }
  };

  v8f acc[2][4];
#pragma unroll
  for (int i = 0; i < 2; ++i)
#pragma unroll
    for (int j = 0; j < 4; ++j) acc[i][j] = zero8();

  stage(0, 0);
  wait_async0();
  __syncthreads();

  int nchunks = K >> 5;
  for (int ci = 0; ci < nchunks; ++ci) {
    int cur = ci & 1;
    if (ci + 1 < nchunks) stage((ci + 1) << 5, cur ^ 1);

    // ---- A fragments (ISA 7.12.2 16-bit A layout), pure ds_load_b128 ----
    v16bf ah[2], al[2];
#pragma unroll
    for (int mt = 0; mt < 2; ++mt) {
      int row = (wm + mt * 16 + l16) * LDT;
      int ko = lhi * 8;  // lanes 0-15: K 0..7/16..23 ; lanes 16-31: K 8..15/24..31
      ah[mt] = cat8(*(const v8bf*)&sAh[cur][row + ko],
                    *(const v8bf*)&sAh[cur][row + 16 + ko]);
      al[mt] = cat8(*(const v8bf*)&sAl[cur][row + ko],
                    *(const v8bf*)&sAl[cur][row + 16 + ko]);
    }

    // ---- per n-tile: B fragment (k-contiguous rows of Bt) + WMMA ----
#pragma unroll
    for (int nt = 0; nt < 4; ++nt) {
      int row = (wn + nt * 16 + l16) * LDT + lhi * 8;  // lanes 0-15: K 0..7/16..23 ; lanes 16-31: K 8..15/24..31
      v16bf bh = cat8(*(const v8bf*)&sBh[cur][row],
                      *(const v8bf*)&sBh[cur][row + 16]);
      v16bf bl = cat8(*(const v8bf*)&sBl[cur][row],
                      *(const v8bf*)&sBl[cur][row + 16]);
#pragma unroll
      for (int mt = 0; mt < 2; ++mt) {
        acc[mt][nt] = __builtin_amdgcn_wmma_f32_16x16x32_bf16(
            false, al[mt], false, bh, (short)0, acc[mt][nt], false, false);
        acc[mt][nt] = __builtin_amdgcn_wmma_f32_16x16x32_bf16(
            false, ah[mt], false, bl, (short)0, acc[mt][nt], false, false);
        acc[mt][nt] = __builtin_amdgcn_wmma_f32_16x16x32_bf16(
            false, ah[mt], false, bh, (short)0, acc[mt][nt], false, false);
      }
    }

    wait_async0();    // our async copies for chunk ci+1 done
    __syncthreads();  // all waves done reading 'cur' buffer
  }

  // ---- epilogue: bias (+ gelu); C layout: VGPR r -> M = r + 8*lhi ----
#pragma unroll
  for (int mt = 0; mt < 2; ++mt)
#pragma unroll
    for (int nt = 0; nt < 4; ++nt) {
      int n = n0 + wn + nt * 16 + l16;
      float bv = be[n];
      int mbase = m0 + wm + mt * 16 + lhi * 8;
#pragma unroll
      for (int r = 0; r < 8; ++r) {
        float v = acc[mt][nt][r] + bv;
        if (DO_GELU) v = gelu_tanh(v);
        if (SPLIT_OUT) {
          __bf16 hh = (__bf16)v;
          Chi[(size_t)(mbase + r) * N + n] = hh;
          Clo[(size_t)(mbase + r) * N + n] = (__bf16)(v - (float)hh);
        } else {
          Cf[(size_t)(mbase + r) * N + n] = v;
        }
      }
    }
}

// =====================================================================
// 5) Combine: out[s] = wgt1*eo[e1,p1] + wgt2*eo[e2,p2]
// =====================================================================
__global__ __launch_bounds__(256) void combine_kernel(
    const float* __restrict__ eo,
    const int* __restrict__ top1, const int* __restrict__ top2,
    const int* __restrict__ pos1, const int* __restrict__ pos2,
    const float* __restrict__ wgt1, const float* __restrict__ wgt2,
    float* __restrict__ out) {
  int s = blockIdx.x;
  int i = threadIdx.x;
  float4 r = {0.f, 0.f, 0.f, 0.f};
  int p1 = pos1[s];
  if (p1 >= 0) {
    float w = wgt1[s];
    float4 v = ((const float4*)eo)[((size_t)top1[s] * kCAP + p1) * (kD / 4) + i];
    r.x += w * v.x; r.y += w * v.y; r.z += w * v.z; r.w += w * v.w;
  }
  int p2 = pos2[s];
  if (p2 >= 0) {
    float w = wgt2[s];
    float4 v = ((const float4*)eo)[((size_t)top2[s] * kCAP + p2) * (kD / 4) + i];
    r.x += w * v.x; r.y += w * v.y; r.z += w * v.z; r.w += w * v.w;
  }
  ((float4*)out)[(size_t)s * (kD / 4) + i] = r;
}

// =====================================================================
// launcher
// =====================================================================
extern "C" void kernel_launch(void* const* d_in, const int* in_sizes, int n_in,
                              void* d_out, int out_size, void* d_ws, size_t ws_size,
                              hipStream_t stream) {
  (void)in_sizes; (void)n_in; (void)out_size; (void)ws_size;
  const float* inputs = (const float*)d_in[0];
  const float* gw     = (const float*)d_in[1];
  const float* w1     = (const float*)d_in[2];
  const float* b1     = (const float*)d_in[3];
  const float* w2     = (const float*)d_in[4];
  const float* b2     = (const float*)d_in[5];
  float* out = (float*)d_out;

  char* ws = (char*)d_ws;
  size_t off = 0;
  auto take = [&](size_t bytes) -> void* {
    void* p = ws + off;
    off += (bytes + 255) & ~(size_t)255;
    return p;
  };
  float*  gates_all = (float*)take((size_t)kS * kE * 4);
  int*    top1      = (int*)take((size_t)kS * 4);
  int*    top2      = (int*)take((size_t)kS * 4);
  float*  g1v       = (float*)take((size_t)kS * 4);
  float*  g2v       = (float*)take((size_t)kS * 4);
  int*    pos1      = (int*)take((size_t)kS * 4);
  int*    pos2      = (int*)take((size_t)kS * 4);
  float*  wgt1      = (float*)take((size_t)kS * 4);
  float*  wgt2      = (float*)take((size_t)kS * 4);
  int*    slot_tok  = (int*)take((size_t)kE * kCAP * 4);
  __bf16* w1t_hi    = (__bf16*)take((size_t)kE * kD * kH * 2);
  __bf16* w1t_lo    = (__bf16*)take((size_t)kE * kD * kH * 2);
  __bf16* w2t_hi    = (__bf16*)take((size_t)kE * kH * kD * 2);
  __bf16* w2t_lo    = (__bf16*)take((size_t)kE * kH * kD * 2);
  __bf16* disp_hi   = (__bf16*)take((size_t)kE * kCAP * kD * 2);
  __bf16* disp_lo   = (__bf16*)take((size_t)kE * kCAP * kD * 2);
  __bf16* h_hi      = (__bf16*)take((size_t)kE * kCAP * kH * 2);
  __bf16* h_lo      = (__bf16*)take((size_t)kE * kCAP * kH * 2);
  float*  eo        = (float*)take((size_t)kE * kCAP * kD * 4);

  // weight pre-pass: fp32 [K][N] -> bf16 hi/lo [N][K] per expert
  split_transpose_kernel<<<dim3(kD / 32, kH / 32, kE), 256, 0, stream>>>(
      w1, w1t_hi, w1t_lo, kD, kH);
  split_transpose_kernel<<<dim3(kH / 32, kD / 32, kE), 256, 0, stream>>>(
      w2, w2t_hi, w2t_lo, kH, kD);

  gate_kernel<<<kS / 8, 256, 0, stream>>>(inputs, gw, gates_all, top1, top2, g1v, g2v);
  scan_kernel<<<1, 1024, 0, stream>>>(top1, top2, g1v, g2v, gates_all,
                                      pos1, pos2, wgt1, wgt2, slot_tok,
                                      out + (size_t)kS * kD);
  gather_kernel<<<kE * kCAP, 256, 0, stream>>>(inputs, slot_tok, disp_hi, disp_lo);

  gemm_bf16x3_kernel<true, true><<<dim3(kCAP / 128, kH / 128, kE), 256, 0, stream>>>(
      disp_hi, disp_lo, w1t_hi, w1t_lo, b1, nullptr, h_hi, h_lo, kCAP, kH, kD);
  gemm_bf16x3_kernel<false, false><<<dim3(kCAP / 128, kD / 128, kE), 256, 0, stream>>>(
      h_hi, h_lo, w2t_hi, w2t_lo, b2, eo, nullptr, nullptr, kCAP, kD, kH);

  combine_kernel<<<kS, 256, 0, stream>>>(eo, top1, top2, pos1, pos2, wgt1, wgt2, out);
}
